// MultiHeadAttention_5111011082875
// MI455X (gfx1250) — matrix operand order repaired and checked by reading, not compiled, not run on hardware
//
#include <hip/hip_runtime.h>
#include <hip/hip_bf16.h>

#define D_IN   2048
#define D_OUTP 2048
#define SEQ    2048
#define NH     16
#define HD     128
#define BATCH  2
#define BM     64
#define BK     32
#define BN     128

typedef __attribute__((ext_vector_type(8)))  float    v8f;
typedef __attribute__((ext_vector_type(8)))  __bf16   v8bf;
typedef __attribute__((ext_vector_type(16))) __bf16   v16bf;
typedef __attribute__((ext_vector_type(4)))  unsigned u32x4;
typedef __attribute__((ext_vector_type(8)))  int      i32x8;
typedef __attribute__((ext_vector_type(4)))  int      i32x4;

__device__ __forceinline__ __bf16 f2bf(float f) {
  unsigned u = __builtin_bit_cast(unsigned, f);
  u += 0x7FFFu + ((u >> 16) & 1u);           // round-to-nearest-even
  unsigned short h = (unsigned short)(u >> 16);
  return __builtin_bit_cast(__bf16, h);
}

// A-fragment builder: 16-bit A 16x32 layout = two contiguous 16B chunks per lane
__device__ __forceinline__ v16bf make_frag(const __bf16* lo, const __bf16* hi) {
  union { v16bf v; v8bf h[2]; } u;
  u.h[0] = *(const v8bf*)lo;
  u.h[1] = *(const v8bf*)hi;
  return u.v;
}

__device__ __forceinline__ v8f v8f_zero() {
  v8f z;
#pragma unroll
  for (int i = 0; i < 8; ++i) z[i] = 0.0f;
  return z;
}

#define WMMA_BF16(a, b, c) \
  __builtin_amdgcn_wmma_f32_16x16x32_bf16(false, (a), false, (b), (short)0, (c), false, false)

// ---------------------------------------------------------------------------
// Tensor Data Mover: 2D tile load (global -> LDS), data_size = 2 bytes.
// D# packing per CDNA5 ISA.
// Dims/strides in elements.  Caller must be wave-uniform; completion is
// tracked with TENSORcnt (s_wait_tensorcnt).  This toolchain's builtin takes
// the 6-arg form: (g0 u32x4, g1 i32x8, g2 i32x4, g3 i32x4, i32x8, cpol).
// ---------------------------------------------------------------------------
__device__ __forceinline__ void tdm_load_2d_bf16(const void* gaddr, unsigned lds_off,
                                                 unsigned tensor_d0, unsigned tensor_d1,
                                                 unsigned tile_d0, unsigned tile_d1,
                                                 unsigned stride_d0)
{
  unsigned long long ga = (unsigned long long)(uintptr_t)gaddr;
  u32x4 g0;
  g0[0] = 1u;                                              // count=1, user mode
  g0[1] = lds_off;                                         // lds_addr (bytes)
  g0[2] = (unsigned)(ga & 0xFFFFFFFFu);                    // global_addr[31:0]
  g0[3] = (unsigned)((ga >> 32) & 0x1FFFFFFu) | (2u << 30);// addr[56:32] | type=2
  i32x8 g1;
  g1[0] = (int)(1u << 16);                                 // data_size=1 (2B)
  g1[1] = (int)((tensor_d0 & 0xFFFFu) << 16);              // tensor_dim0[15:0]
  g1[2] = (int)(((tensor_d0 >> 16) & 0xFFFFu) |
                ((tensor_d1 & 0xFFFFu) << 16));            // dim0[31:16]|dim1[15:0]
  g1[3] = (int)(((tensor_d1 >> 16) & 0xFFFFu) |
                ((tile_d0 & 0xFFFFu) << 16));              // dim1[31:16]|tile0
  g1[4] = (int)(tile_d1 & 0xFFFFu);                        // tile1, tile2=0
  g1[5] = (int)stride_d0;                                  // dim0_stride[31:0]
  g1[6] = 0;                                               // stride hi, dim1_stride
  g1[7] = 0;
  i32x4 z4;
  z4[0] = 0; z4[1] = 0; z4[2] = 0; z4[3] = 0;              // 2-D: groups 2/3 zero
  i32x8 z8;
#pragma unroll
  for (int i = 0; i < 8; ++i) z8[i] = 0;
  __builtin_amdgcn_tensor_load_to_lds(g0, g1, z4, z4, z8, 0);
}

// ---------------------------------------------------------------------------
// Fused GEMM (+ optional RoPE epilogue).  Tile: 64 rows x 128 cols, K-step 32.
// 8 waves: 4 row-groups x 2 column-groups; column group cg owns columns
// {cg*32..cg*32+31} U {cg*32+64..cg*32+95} so RoPE pairs (d, d+64) stay
// inside one wave's accumulators.
// ---------------------------------------------------------------------------
template <bool A_BF16, bool DO_ROPE, bool OUT_F32>
__global__ __launch_bounds__(256)
void gemm_rope_kernel(const float* __restrict__ Af, const __bf16* __restrict__ Ab,
                      const float* __restrict__ W,
                      __bf16* __restrict__ outB, float* __restrict__ outF)
{
  __shared__ __bf16 Ash[BM][BK];     // 64 x 32 bf16 (4 KB)
  __shared__ __bf16 Bsh[BN * BK];    // transposed [n][k], 128 x 32 (8 KB)

  const int tid  = threadIdx.x;
  const int w    = tid >> 5;
  const int lane = tid & 31;
  const int ln   = lane & 15;
  const int hi   = lane >> 4;
  const int m0   = (w & 3) * 16;
  const int ncg  = (w >> 2) * 32;
  const int sub_n[4] = {ncg, ncg + 16, ncg + 64, ncg + 80};

  const int rowTile = blockIdx.x;
  const int n0      = blockIdx.y * BN;

  v8f acc[4];
#pragma unroll
  for (int j = 0; j < 4; ++j) acc[j] = v8f_zero();

  const int a_row = tid >> 2;        // 0..63
  const int a_k   = (tid & 3) * 8;   // 0,8,16,24
  const int b_k   = tid >> 3;        // 0..31
  const int b_n   = (tid & 7) * 16;  // 0..112

  for (int bk = 0; bk < D_IN / BK; ++bk) {
    const int k0 = bk * BK;
    if (A_BF16) {
      v8bf a = *(const v8bf*)(Ab + (size_t)(rowTile * BM + a_row) * D_IN + k0 + a_k);
      *(v8bf*)&Ash[a_row][a_k] = a;
    } else {
      const float* src = Af + (size_t)(rowTile * BM + a_row) * D_IN + k0 + a_k;
      v8bf a;
#pragma unroll
      for (int i = 0; i < 8; ++i) a[i] = f2bf(src[i]);
      *(v8bf*)&Ash[a_row][a_k] = a;
    }
    {
      const float* src = W + (size_t)(k0 + b_k) * D_OUTP + n0 + b_n;
#pragma unroll
      for (int i = 0; i < 16; ++i)
        Bsh[(b_n + i) * BK + b_k] = f2bf(src[i]);
    }
    // prefetch next K-tile (lowers to global_prefetch_b8)
    if (bk + 1 < D_IN / BK) {
      __builtin_prefetch(W + (size_t)(k0 + BK + b_k) * D_OUTP + n0 + b_n, 0, 0);
      if (!A_BF16)
        __builtin_prefetch(Af + (size_t)(rowTile * BM + a_row) * D_IN + k0 + BK + a_k, 0, 0);
    }
    __syncthreads();

    v16bf af = make_frag(&Ash[m0 + ln][hi * 16], &Ash[m0 + ln][hi * 16 + 8]);
#pragma unroll
    for (int j = 0; j < 4; ++j) {
      v16bf bfg = *(const v16bf*)&Bsh[(sub_n[j] + ln) * BK + hi * 16];
      acc[j] = WMMA_BF16(af, bfg, acc[j]);
    }
    __syncthreads();
  }

  // ---- epilogue ----
#pragma unroll
  for (int r = 0; r < 8; ++r) {
    const int row_l = m0 + r + 8 * hi;
    const int row_g = rowTile * BM + row_l;
    const int bidx  = row_g >> 11;          // / SEQ
    const int spos  = row_g & (SEQ - 1);
    if (DO_ROPE) {
      const int h = blockIdx.y;             // grid.y == head for QKV
      __bf16* dst = outB + ((size_t)(bidx * NH + h) * SEQ + spos) * HD;
#pragma unroll
      for (int j = 0; j < 2; ++j) {
        const int f = sub_n[j] + ln;        // rotation-pair index, 0..63
        const float invf = __powf(10000.0f, -(float)f * (1.0f / 64.0f));
        float sn, cs;
        __sincosf((float)spos * invf, &sn, &cs);
        const float x1 = acc[j][r];
        const float x2 = acc[j + 2][r];
        dst[f]      = f2bf(x1 * cs - x2 * sn);
        dst[f + 64] = f2bf(x2 * cs + x1 * sn);
      }
    } else if (!OUT_F32) {
      const int h = blockIdx.y;
      __bf16* dst = outB + ((size_t)(bidx * NH + h) * SEQ + spos) * HD;
#pragma unroll
      for (int j = 0; j < 4; ++j) dst[sub_n[j] + ln] = f2bf(acc[j][r]);
    } else {
#pragma unroll
      for (int j = 0; j < 4; ++j)
        outF[(size_t)row_g * D_OUTP + n0 + sub_n[j] + ln] = acc[j][r];
    }
  }
}

// ---------------------------------------------------------------------------
// Flash attention: 64 query rows per block (4 waves x 16 rows), online softmax,
// QK^T and P*V via bf16 WMMA.  Q and K tiles are DMA'd into LDS by the Tensor
// Data Mover (wave 0 issues, TENSORcnt-synchronized); V is transposed manually
// into [d][s] so P*V B-fragments are contiguous.  P staged per-wave in LDS to
// convert C-layout -> A-layout (same-wave DS ordering, no barrier needed).
// ---------------------------------------------------------------------------
__global__ __launch_bounds__(128)
void flash_attn_kernel(const __bf16* __restrict__ Qb, const __bf16* __restrict__ Kb,
                       const __bf16* __restrict__ Vb, __bf16* __restrict__ Ctx)
{
  __shared__ __bf16 Qs[64][HD];     // 16 KB
  __shared__ __bf16 Ks[64][HD];     // 16 KB
  __shared__ __bf16 Vt[HD][64];     // 16 KB, transposed
  __shared__ __bf16 Ps[4][16][64];  //  8 KB, per-wave P staging

  const int tid  = threadIdx.x;
  const int w    = tid >> 5;
  const int lane = tid & 31;
  const int ln   = lane & 15;
  const int hi   = lane >> 4;
  const int m0   = w * 16;
  const int qb = blockIdx.x, h = blockIdx.y, b = blockIdx.z;
  const size_t base = ((size_t)(b * NH + h)) * SEQ * HD;

  // ---- Q tile via TDM: 64 rows x 128 cols, row stride HD ----
  if (w == 0) {
    tdm_load_2d_bf16(Qb + base + (size_t)(qb * 64) * HD,
                     (unsigned)(uintptr_t)&Qs[0][0],
                     /*tensor_d0=*/HD, /*tensor_d1=*/SEQ,
                     /*tile_d0=*/HD, /*tile_d1=*/64, /*stride_d0=*/HD);
  }
  __builtin_amdgcn_s_wait_tensorcnt(0);

  float mi[8], li[8];
  v8f o[8];
#pragma unroll
  for (int r = 0; r < 8; ++r) { mi[r] = -__builtin_inff(); li[r] = 0.0f; }
#pragma unroll
  for (int dt = 0; dt < 8; ++dt) o[dt] = v8f_zero();

  for (int kb = 0; kb <= qb; ++kb) {
    __syncthreads();
    // ---- K tile via TDM (layout-preserving 2D copy) ----
    if (w == 0) {
      tdm_load_2d_bf16(Kb + base + (size_t)(kb * 64) * HD,
                       (unsigned)(uintptr_t)&Ks[0][0],
                       HD, SEQ, HD, 64, HD);
    }
    // ---- V tile: manual transposing copy ----
    {
      const int row = tid >> 1;
      const int d0  = (tid & 1) * 64;
#pragma unroll
      for (int i = 0; i < 8; ++i) {
        v8bf vv = *(const v8bf*)(Vb + base + (size_t)(kb * 64 + row) * HD + d0 + i * 8);
#pragma unroll
        for (int e = 0; e < 8; ++e) Vt[d0 + i * 8 + e][row] = vv[e];
      }
    }
    __builtin_amdgcn_s_wait_tensorcnt(0);
    __syncthreads();

    // ---- S = Q @ K^T, 16x64 per wave ----
    v8f sa[4];
#pragma unroll
    for (int nt = 0; nt < 4; ++nt) sa[nt] = v8f_zero();
#pragma unroll
    for (int kc = 0; kc < HD; kc += 32) {
      v16bf qf = make_frag(&Qs[m0 + ln][kc + hi * 16], &Qs[m0 + ln][kc + hi * 16 + 8]);
#pragma unroll
      for (int nt = 0; nt < 4; ++nt) {
        v16bf kf = *(const v16bf*)&Ks[nt * 16 + ln][kc + hi * 16];
        sa[nt] = WMMA_BF16(qf, kf, sa[nt]);
      }
    }

    // ---- scale + causal mask (diagonal block only) ----
    const float sc = 0.08838834764831845f;   // 1/sqrt(128)
    const bool diag = (kb == qb);
#pragma unroll
    for (int nt = 0; nt < 4; ++nt) {
#pragma unroll
      for (int r = 0; r < 8; ++r) {
        float v = sa[nt][r] * sc;
        if (diag) {
          const int kg = nt * 16 + ln;
          const int qg = m0 + r + 8 * hi;
          if (kg > qg) v = -__builtin_inff();
        }
        sa[nt][r] = v;
      }
    }

    // ---- online softmax: rows reduce across the 16-lane half ----
    float scale_r[8];
#pragma unroll
    for (int r = 0; r < 8; ++r) {
      float rm = sa[0][r];
#pragma unroll
      for (int nt = 1; nt < 4; ++nt) rm = fmaxf(rm, sa[nt][r]);
#pragma unroll
      for (int off = 1; off < 16; off <<= 1)
        rm = fmaxf(rm, __shfl_xor(rm, off, 16));
      const float mnew = fmaxf(mi[r], rm);
      scale_r[r] = __expf(mi[r] - mnew);
      float sum = 0.0f;
#pragma unroll
      for (int nt = 0; nt < 4; ++nt) {
        float p = __expf(sa[nt][r] - mnew);
        sa[nt][r] = p;
        sum += p;
      }
#pragma unroll
      for (int off = 1; off < 16; off <<= 1)
        sum += __shfl_xor(sum, off, 16);
      li[r] = li[r] * scale_r[r] + sum;
      mi[r] = mnew;
    }

#pragma unroll
    for (int dt = 0; dt < 8; ++dt)
#pragma unroll
      for (int r = 0; r < 8; ++r) o[dt][r] *= scale_r[r];

    // ---- stage P (C-layout -> A-layout via per-wave LDS) ----
#pragma unroll
    for (int nt = 0; nt < 4; ++nt)
#pragma unroll
      for (int r = 0; r < 8; ++r)
        Ps[w][r + 8 * hi][nt * 16 + ln] = f2bf(sa[nt][r]);

    // ---- O += P @ V ----
#pragma unroll
    for (int kc2 = 0; kc2 < 64; kc2 += 32) {
      v16bf pf = make_frag(&Ps[w][ln][kc2 + hi * 16], &Ps[w][ln][kc2 + hi * 16 + 8])
#pragma unroll
      for (int dt = 0; dt < 8; ++dt) {
        v16bf vf = *(const v16bf*)&Vt[dt * 16 + ln][kc2 + hi * 16];
        o[dt] = WMMA_BF16(pf, vf, o[dt]);
      }
    }
  }

  // ---- normalize, write ctx in merged-head [b][s][h*128+d] bf16 layout ----
#pragma unroll
  for (int r = 0; r < 8; ++r) {
    const int s = qb * 64 + m0 + r + 8 * hi;
    const float inv_l = 1.0f / li[r];
    __bf16* dst = Ctx + ((size_t)b * SEQ + s) * D_OUTP + h * HD;
#pragma unroll
    for (int dt = 0; dt < 8; ++dt)
      dst[dt * 16 + ln] = f2bf(o[dt][r] * inv_l);
  }
}

// ---------------------------------------------------------------------------
extern "C" void kernel_launch(void* const* d_in, const int* in_sizes, int n_in,
                              void* d_out, int out_size, void* d_ws, size_t ws_size,
                              hipStream_t stream) {
  (void)in_sizes; (void)n_in; (void)out_size; (void)ws_size;
  const float* x  = (const float*)d_in[0];
  const float* Wq = (const float*)d_in[1];
  const float* Wk = (const float*)d_in[2];
  const float* Wv = (const float*)d_in[3];
  const float* Wo = (const float*)d_in[4];
  float* out = (float*)d_out;

  const size_t per = (size_t)BATCH * NH * SEQ * HD;   // 8.39M bf16 elems each
  __bf16* Qb  = (__bf16*)d_ws;
  __bf16* Kb  = Qb + per;
  __bf16* Vb  = Kb + per;
  __bf16* Ctx = Vb + per;                             // total 64 MB of d_ws

  dim3 gGemm(BATCH * SEQ / BM, D_OUTP / BN, 1);       // (64, 16)
  dim3 bGemm(256);
  gemm_rope_kernel<false, true,  false><<<gGemm, bGemm, 0, stream>>>(x, nullptr, Wq, Qb, nullptr);
  gemm_rope_kernel<false, true,  false><<<gGemm, bGemm, 0, stream>>>(x, nullptr, Wk, Kb, nullptr);
  gemm_rope_kernel<false, false, false><<<gGemm, bGemm, 0, stream>>>(x, nullptr, Wv, Vb, nullptr);

  dim3 gAttn(SEQ / 64, NH, BATCH);
  flash_attn_kernel<<<gAttn, dim3(128), 0, stream>>>(Qb, Kb, Vb, Ctx);

  gemm_rope_kernel<true, false, true><<<gGemm, bGemm, 0, stream>>>(nullptr, Ctx, Wo, nullptr, out);
}
